// OptimizedMultiDimensionalLSTM_27530740367825
// MI455X (gfx1250) — matrix operand order repaired and checked by reading, not compiled, not run on hardware
//
#include <hip/hip_runtime.h>
#include <hip/hip_bf16.h>

typedef _Float16 v16h __attribute__((ext_vector_type(16)));
typedef _Float16 v8h  __attribute__((ext_vector_type(8)));
typedef float    v8f  __attribute__((ext_vector_type(8)));

// Problem dims
static constexpr int Bdim = 32;    // batch
static constexpr int Hn   = 32;    // grid rows
static constexpr int Wn   = 32;    // grid cols
static constexpr int Fdim = 256;   // features per cell
static constexpr int Rdim = 512;   // rnn size
static constexpr int Ktot = 1280;  // F + 2R
static constexpr int LSTRIDE = 1288; // LDS row stride in halfs (1280 + 8 pad)

// Branch-free fast sigmoid / tanh on v_exp_f32 + v_rcp_f32.
__device__ __forceinline__ float sigf(float x) {
  return __builtin_amdgcn_rcpf(1.0f + __expf(-x));
}
__device__ __forceinline__ float tanh_fast(float x) {
  return 2.0f * sigf(2.0f * x) - 1.0f;
}

// ---------------------------------------------------------------------------
// Pack x [B,C,128,128] f32 -> Xg [cell(1024)][b(32)][k(256)] f16
// ---------------------------------------------------------------------------
__global__ void pack_x_kernel(const float* __restrict__ x, _Float16* __restrict__ Xg) {
  int e = blockIdx.x * blockDim.x + threadIdx.x;  // 1024*32*256 total
  int cell = e >> 13;
  int rem  = e & 8191;
  int b    = rem >> 8;
  int k    = rem & 255;
  int i = cell >> 5, j = cell & 31;
  int flat = i * 8192 + j * 256 + k;   // raw reshape index into XD*YD*C
  int xd = flat >> 11;
  int yd = (flat >> 4) & 127;
  int c  = flat & 15;
  float v = x[(((b * 16 + c) * 128 + xd) * 128) + yd];
  Xg[(size_t)e] = (_Float16)v;
}

// ---------------------------------------------------------------------------
// Pack W [1280,2560] f32 -> Wt [n(2560)][k(1280)] f16 (transposed)
// ---------------------------------------------------------------------------
__global__ void pack_w_kernel(const float* __restrict__ W, _Float16* __restrict__ Wt) {
  int e = blockIdx.x * blockDim.x + threadIdx.x;  // 1280*2560 total
  int k = e / 2560;
  int n = e % 2560;
  Wt[(size_t)n * Ktot + k] = (_Float16)W[e];
}

// Zero the dummy "missing neighbor" slot (64 KB = 16384 f32 = one C slot;
// its first 32 KB doubles as a zero f16 H slot).
__global__ void zero_kernel(float* __restrict__ z) {
  z[blockIdx.x * blockDim.x + threadIdx.x] = 0.0f;
}

// ---------------------------------------------------------------------------
// grid = (ncells, 2): one cell per blockIdx.x, slice-half per blockIdx.y.
// 16 waves/block; wave handles exactly one r-slice s = wave + 16*blockIdx.y,
// owning the 5 gate N-tiles x 2 M-tiles (80 accum VGPRs) for that slice.
// ---------------------------------------------------------------------------
__global__ __launch_bounds__(512) void mdlstm_diag_kernel(
    int d, int i0,
    const _Float16* __restrict__ Xg,
    const _Float16* __restrict__ Wt,
    const float*    __restrict__ bias,
    const _Float16* __restrict__ Hprev, _Float16* __restrict__ Hcur,
    const float*    __restrict__ Cprev, float*    __restrict__ Ccur,
    const float*    __restrict__ Zero,   // 64 KB of zeros
    float* __restrict__ out)
{
  __shared__ _Float16 Alds[Bdim * LSTRIDE];   // ~80.5 KB

  const int tid = threadIdx.x;
  const int i = i0 + blockIdx.x;
  const int j = d - i;
  const int cell = i * Wn + j;
  constexpr size_t HS = (size_t)Bdim * Rdim;   // elements per H/C slot

  // Uniform pointer selects: missing neighbors read the zero slot.
  const _Float16* hup  = (i > 0) ? (Hprev + (size_t)(i - 1) * HS) : (const _Float16*)Zero;
  const _Float16* hlf  = (j > 0) ? (Hprev + (size_t)i * HS)       : (const _Float16*)Zero;
  const float*    cupb = (i > 0) ? (Cprev + (size_t)(i - 1) * HS) : Zero;
  const float*    clfb = (j > 0) ? (Cprev + (size_t)i * HS)       : Zero;

  // ---- stage A = [Xg | h_up | h_left] into LDS (straight-line copies) ----
  {
    const uint4* src = (const uint4*)(Xg + (size_t)cell * (Bdim * Fdim));
    for (int c = tid; c < (Bdim * Fdim) / 8; c += blockDim.x) {
      int row = c >> 5;
      int col = (c & 31) << 3;
      *(uint4*)(&Alds[row * LSTRIDE + col]) = src[c];
    }
  }
  {
    const uint4* src = (const uint4*)hup;
    for (int c = tid; c < (Bdim * Rdim) / 8; c += blockDim.x) {
      int row = c >> 6;
      int col = Fdim + ((c & 63) << 3);
      *(uint4*)(&Alds[row * LSTRIDE + col]) = src[c];
    }
  }
  {
    const uint4* src = (const uint4*)hlf;
    for (int c = tid; c < (Bdim * Rdim) / 8; c += blockDim.x) {
      int row = c >> 6;
      int col = Fdim + Rdim + ((c & 63) << 3);
      *(uint4*)(&Alds[row * LSTRIDE + col]) = src[c];
    }
  }
  __syncthreads();

  const int lane  = tid & 31;
  const int wave  = tid >> 5;
  const int nlo   = lane & 15;       // N (or M row) within tile
  const int hi    = lane >> 4;       // lane half select
  const int khalfA = hi * 16;        // A: lanes>=16 hold K 16..31
  const int kbaseB = hi * 16;        // B: lanes>=16 hold K 16..31

  constexpr size_t GSTRIDE = (size_t)Rdim * Ktot;  // per-gate Wt column stride

  const int s = wave + 16 * blockIdx.y;            // this wave's r-slice
  const int rcol = s * 16 + nlo;                   // r in [0,512)

  v8f acc[5][2];
  #pragma unroll
  for (int g = 0; g < 5; ++g) {
    float bv = bias[g * Rdim + rcol];              // fold bias into C init
    #pragma unroll
    for (int e = 0; e < 8; ++e) { acc[g][0][e] = bv; acc[g][1][e] = bv; }
  }

  const _Float16* wbase  = Wt + (size_t)rcol * Ktot + kbaseB;
  const _Float16* a0base = &Alds[nlo * LSTRIDE + khalfA];
  const _Float16* a1base = &Alds[(16 + nlo) * LSTRIDE + khalfA];

  #pragma unroll 1
  for (int kb = 0; kb < Ktot / 32; ++kb) {
    const int k0 = kb * 32;

    union { v16h v; v8h h[2]; } ua0, ua1;
    ua0.h[0] = *(const v8h*)(a0base + k0);
    ua0.h[1] = *(const v8h*)(a0base + k0 + 8);
    ua1.h[0] = *(const v8h*)(a1base + k0);
    ua1.h[1] = *(const v8h*)(a1base + k0 + 8);

    v16h bm[5];
    #pragma unroll
    for (int g = 0; g < 5; ++g)
      bm[g] = *(const v16h*)(wbase + g * GSTRIDE + k0);

    #pragma unroll
    for (int g = 0; g < 5; ++g) {
      acc[g][0] = __builtin_amdgcn_wmma_f32_16x16x32_f16(
          false, ua0.v, false, bm[g], (short)0, acc[g][0], false, false);
      acc[g][1] = __builtin_amdgcn_wmma_f32_16x16x32_f16(
          false, ua1.v, false, bm[g], (short)0, acc[g][1], false, false);
    }
  }

  // ---- gates: i,j,f1,f2,o live in registers in D layout; branch-free ----
  #pragma unroll
  for (int mt = 0; mt < 2; ++mt) {
    #pragma unroll
    for (int r = 0; r < 8; ++r) {
      const int bidx = mt * 16 + hi * 8 + r;       // batch row (D layout)
      const size_t soff = (size_t)bidx * Rdim + rcol;
      float zi  = acc[0][mt][r];
      float zj  = acc[1][mt][r];
      float zf1 = acc[2][mt][r];
      float zf2 = acc[3][mt][r];
      float zo  = acc[4][mt][r];
      float cu = cupb[soff];
      float cl = clfb[soff];
      float nc = cu * sigf(zf1) + cl * sigf(zf2) + sigf(zi) * tanh_fast(zj);
      float nh = tanh_fast(nc) * sigf(zo);
      Ccur[(size_t)i * HS + soff] = nc;
      Hcur[(size_t)i * HS + soff] = (_Float16)nh;
      out[(((size_t)bidx * Hn + i) * Wn + j) * Rdim + rcol] = nh;
    }
  }
}

// ---------------------------------------------------------------------------
extern "C" void kernel_launch(void* const* d_in, const int* in_sizes, int n_in,
                              void* d_out, int out_size, void* d_ws, size_t ws_size,
                              hipStream_t stream) {
  (void)in_sizes; (void)n_in; (void)out_size; (void)ws_size;
  const float* x  = (const float*)d_in[0];
  const float* W  = (const float*)d_in[1];
  const float* bs = (const float*)d_in[2];
  float* out = (float*)d_out;
  char* ws = (char*)d_ws;

  size_t off = 0;
  auto take = [&](size_t bytes) -> char* {
    char* p = ws + off;
    off += (bytes + 255) & ~(size_t)255;
    return p;
  };
  _Float16* Xg = (_Float16*)take((size_t)1024 * Bdim * Fdim * sizeof(_Float16)); // 16 MB
  _Float16* Wt = (_Float16*)take((size_t)2560 * Ktot * sizeof(_Float16));        // 6.5 MB
  const size_t HC = (size_t)Hn * Bdim * Rdim;   // one diagonal buffer (32 slots)
  _Float16* Hpp = (_Float16*)take(2 * HC * sizeof(_Float16));                    // 2 MB
  float*    Cpp = (float*)take(2 * HC * sizeof(float));                          // 4 MB
  float*    Zb  = (float*)take((size_t)Bdim * Rdim * sizeof(float));             // 64 KB zeros

  pack_x_kernel<<<dim3(32768), dim3(256), 0, stream>>>(x, Xg);
  pack_w_kernel<<<dim3(12800), dim3(256), 0, stream>>>(W, Wt);
  zero_kernel<<<dim3(64), dim3(256), 0, stream>>>(Zb);

  // Wavefront over 63 diagonals; inter-diagonal dependency via stream order.
  for (int d = 0; d < 63; ++d) {
    int i0 = (d > 31) ? (d - 31) : 0;
    int i1 = (d < 31) ? d : 31;
    int ncells = i1 - i0 + 1;
    int cur = d & 1, prev = cur ^ 1;
    mdlstm_diag_kernel<<<dim3(ncells, 2), dim3(512), 0, stream>>>(
        d, i0, Xg, Wt, bs,
        Hpp + (size_t)prev * HC, Hpp + (size_t)cur * HC,
        Cpp + (size_t)prev * HC, Cpp + (size_t)cur * HC,
        Zb, out);
  }
}
